// AdapterModule_46050639348236
// MI455X (gfx1250) — matrix operand order repaired and checked by reading, not compiled, not run on hardware
//
#include <hip/hip_runtime.h>
#include <hip/hip_bf16.h>

// ---------------------------------------------------------------------------
// Kron-adapter fused as: W = sum_i softmax(gate)_i * (B_i (x) A_i)  [128x2048]
// out = LayerNorm(SiLU(x @ W^T))  via split-bf16 (3x) WMMA GEMM, fp32 accum.
// ---------------------------------------------------------------------------

typedef __attribute__((ext_vector_type(16))) __bf16 v16bf;
typedef __attribute__((ext_vector_type(8)))  float  v8f;

#define N_IN   2048
#define M_OUT  128
#define KB     256            // K-chunk staged in LDS
#define KROW   (KB + 8)       // padded LDS row (ushorts) -> avoids bank conflicts
#define MTILE  64             // tokens per block
#define LN_EPS 1e-5f

__device__ __forceinline__ unsigned short f2bf(float f) {
    unsigned u = __float_as_uint(f);
    unsigned r = u + 0x7FFFu + ((u >> 16) & 1u);   // round-to-nearest-even
    return (unsigned short)(r >> 16);
}
__device__ __forceinline__ float bf2f(unsigned short h) {
    return __uint_as_float(((unsigned)h) << 16);
}

struct FactorPtrs {
    const float* A[10];
    const float* B[10];
    const float* gate;
};

// ---- Kernel 1: build W_hi / W_lo (bf16) from the 10 Kronecker factors ------
__global__ void build_w_kernel(FactorPtrs fp,
                               unsigned short* __restrict__ Whi,
                               unsigned short* __restrict__ Wlo) {
    const int MA[10]  = {8,16,4,2,32, 8,16,4,2,32};
    const int NA[10]  = {32,16,64,128,8, 32,16,64,128,8};
    const int NBv[10] = {64,128,32,16,256, 64,128,32,16,256};

    // softmax over the 10 gates (tiny, recomputed per thread)
    float g[10], a[10], mx = -1e30f;
    #pragma unroll
    for (int i = 0; i < 10; ++i) { g[i] = fp.gate[i]; mx = fmaxf(mx, g[i]); }
    float s = 0.f;
    #pragma unroll
    for (int i = 0; i < 10; ++i) { a[i] = __expf(g[i] - mx); s += a[i]; }
    const float inv = 1.f / s;

    const int r = blockIdx.x;                       // output row 0..127
    for (int c = threadIdx.x; c < N_IN; c += blockDim.x) {
        float w = 0.f;
        #pragma unroll
        for (int i = 0; i < 10; ++i) {
            const int ja = r % MA[i], jb = r / MA[i];
            const int l  = c % NA[i], k  = c / NA[i];
            w += (a[i] * inv) * fp.B[i][jb * NBv[i] + k] * fp.A[i][ja * NA[i] + l];
        }
        const unsigned short hi = f2bf(w);
        Whi[r * N_IN + c] = hi;
        Wlo[r * N_IN + c] = f2bf(w - bf2f(hi));
    }
}

// ---- Kernel 2: split-bf16 WMMA GEMM + SiLU + LayerNorm ---------------------
__global__ __launch_bounds__(256)
void kron_gemm_ln_kernel(const float* __restrict__ x,
                         const unsigned short* __restrict__ Whi,
                         const unsigned short* __restrict__ Wlo,
                         const float* __restrict__ gamma,
                         const float* __restrict__ beta,
                         float* __restrict__ out) {
    __shared__ __align__(16) unsigned short ldsX[2 * MTILE * KROW]; // hi|lo planes

    const int  tid  = threadIdx.x;
    const int  wave = tid >> 5;         // 8 waves -> 8 N-tiles of 16
    const int  lane = tid & 31;
    const int  half = lane >> 4;
    const int  lrow = lane & 15;
    const long t0   = (long)blockIdx.x * MTILE;

    v8f acc[4];
    #pragma unroll
    for (int mt = 0; mt < 4; ++mt) acc[mt] = (v8f){0,0,0,0,0,0,0,0};

    const int nBase = wave * 16;
    // B frag: lane holds column n=lrow, K-rows (half*16 + j), j=0..15 contiguous
    const unsigned short* WhiRow = Whi + (size_t)(nBase + lrow) * N_IN + half * 16;
    const unsigned short* WloRow = Wlo + (size_t)(nBase + lrow) * N_IN + half * 16;

    for (int kc = 0; kc < N_IN; kc += KB) {
        // ---- stage 64x256 fp32 chunk of x into LDS as bf16 hi/lo -----------
        #pragma unroll
        for (int it = 0; it < 16; ++it) {
            const int e4    = tid + it * 256;  // float4 index over chunk
            const int token = e4 >> 6;         // 64 float4 per token row
            const int kq    = e4 & 63;
            const float4 v  = *(const float4*)(x + (t0 + token) * N_IN + kc + kq * 4);
            const unsigned short h0 = f2bf(v.x), h1 = f2bf(v.y),
                                 h2 = f2bf(v.z), h3 = f2bf(v.w);
            uint2 hp, lp;
            hp.x = (unsigned)h0 | ((unsigned)h1 << 16);
            hp.y = (unsigned)h2 | ((unsigned)h3 << 16);
            lp.x = (unsigned)f2bf(v.x - bf2f(h0)) | ((unsigned)f2bf(v.y - bf2f(h1)) << 16);
            lp.y = (unsigned)f2bf(v.z - bf2f(h2)) | ((unsigned)f2bf(v.w - bf2f(h3)) << 16);
            *(uint2*)(&ldsX[token * KROW + kq * 4])                = hp;
            *(uint2*)(&ldsX[MTILE * KROW + token * KROW + kq * 4]) = lp;
        }
        __syncthreads();

        for (int kk = 0; kk < KB; kk += 32) {
            union { v16bf v; uint4 q[2]; } bh, bl;
            const unsigned short* ph = WhiRow + kc + kk;
            const unsigned short* pl = WloRow + kc + kk;
            bh.q[0] = *(const uint4*)(ph);     bh.q[1] = *(const uint4*)(ph + 8);
            bl.q[0] = *(const uint4*)(pl);     bl.q[1] = *(const uint4*)(pl + 8);
            #pragma unroll
            for (int mt = 0; mt < 4; ++mt) {
                // A frag (16x32 bf16): lane M=lrow; j0..15 -> K=half*16+j
                // (contiguous, matching B frag)
                union { v16bf v; uint4 q[2]; } ah, al;
                const unsigned short* rx = &ldsX[(mt * 16 + lrow) * KROW + kk];
                const unsigned short* rl = rx + MTILE * KROW;
                ah.q[0] = *(const uint4*)(rx + half * 16);
                ah.q[1] = *(const uint4*)(rx + half * 16 + 8);
                al.q[0] = *(const uint4*)(rl + half * 16);
                al.q[1] = *(const uint4*)(rl + half * 16 + 8);
                acc[mt] = __builtin_amdgcn_wmma_f32_16x16x32_bf16(
                    false, ah.v, false, bh.v, (short)0, acc[mt], false, false);
                acc[mt] = __builtin_amdgcn_wmma_f32_16x16x32_bf16(
                    false, ah.v, false, bl.v, (short)0, acc[mt], false, false);
                acc[mt] = __builtin_amdgcn_wmma_f32_16x16x32_bf16(
                    false, al.v, false, bh.v, (short)0, acc[mt], false, false);
            }
        }
        __syncthreads();
    }

    // ---- epilogue: SiLU -> LDS, per-token LayerNorm, store fp32 ------------
    float* ldsAct = (float*)ldsX;                 // reuse (barrier above)
    const int AS = M_OUT + 4;                     // padded stride
    #pragma unroll
    for (int mt = 0; mt < 4; ++mt) {
        #pragma unroll
        for (int r = 0; r < 8; ++r) {
            const int tm = mt * 16 + half * 8 + r;   // C/D: M = r + 8*(lane/16)
            const float v = acc[mt][r];
            ldsAct[tm * AS + nBase + lrow] = v / (1.f + __expf(-v));  // SiLU
        }
    }
    __syncthreads();

    float gma[4], bta[4];
    #pragma unroll
    for (int j = 0; j < 4; ++j) { gma[j] = gamma[lane + 32 * j]; bta[j] = beta[lane + 32 * j]; }

    for (int q = 0; q < 8; ++q) {
        const int tm = wave * 8 + q;
        float av[4], s1 = 0.f, s2 = 0.f;
        #pragma unroll
        for (int j = 0; j < 4; ++j) {
            av[j] = ldsAct[tm * AS + lane + 32 * j];
            s1 += av[j]; s2 += av[j] * av[j];
        }
        #pragma unroll
        for (int off = 16; off >= 1; off >>= 1) {
            s1 += __shfl_xor(s1, off, 32);
            s2 += __shfl_xor(s2, off, 32);
        }
        const float mean = s1 * (1.f / 128.f);
        const float var  = s2 * (1.f / 128.f) - mean * mean;
        const float rstd = rsqrtf(var + LN_EPS);
        float* orow = out + (t0 + tm) * M_OUT;
        #pragma unroll
        for (int j = 0; j < 4; ++j)
            orow[lane + 32 * j] = (av[j] - mean) * rstd * gma[j] + bta[j];
    }
}

// ---------------------------------------------------------------------------
extern "C" void kernel_launch(void* const* d_in, const int* in_sizes, int n_in,
                              void* d_out, int out_size, void* d_ws, size_t ws_size,
                              hipStream_t stream) {
    // dict order: x, A_factors[0..9], B_factors[0..9], gate, ln_gamma, ln_beta
    const float* x = (const float*)d_in[0];
    FactorPtrs fp;
    for (int i = 0; i < 10; ++i) {
        fp.A[i] = (const float*)d_in[1 + i];
        fp.B[i] = (const float*)d_in[11 + i];
    }
    fp.gate = (const float*)d_in[21];
    const float* gamma = (const float*)d_in[22];
    const float* beta  = (const float*)d_in[23];

    unsigned short* Whi = (unsigned short*)d_ws;
    unsigned short* Wlo = Whi + (size_t)M_OUT * N_IN;

    const long T = (long)in_sizes[0] / N_IN;     // 32768, divisible by MTILE

    build_w_kernel<<<M_OUT, 256, 0, stream>>>(fp, Whi, Wlo);
    kron_gemm_ln_kernel<<<(int)(T / MTILE), 256, 0, stream>>>(
        x, Whi, Wlo, gamma, beta, (float*)d_out);
}
